// AdaptiveGRU_71700184039677
// MI455X (gfx1250) — matrix operand order repaired and checked by reading, not compiled, not run on hardware
//
#include <hip/hip_runtime.h>
#include <hip/hip_bf16.h>
#include <math.h>

typedef __attribute__((ext_vector_type(16))) __bf16 v16bf;
typedef __attribute__((ext_vector_type(8)))  float  v8f;

namespace {

constexpr int kB = 64, kT = 512, kF = 256, kH = 1024, kOUT = 256;
constexpr int kN1 = 3 * kH;                       // 3072 gate columns
constexpr float kTEMP = 0.05f;
constexpr float kEPS  = 1e-20f;

constexpr int NBLK  = 96;
constexpr int NTHR  = 256;
constexpr int NWAVE = NBLK * (NTHR / 32);         // 768 waves

// d_out layout: outputs [T,B,OUT] fp32, num_selections (1), weights [T,B,F]
constexpr size_t NSEL_OFF = (size_t)kT * kB * kOUT;   // 8388608
constexpr size_t WSEL_OUT_OFF = NSEL_OFF + 1;

// workspace byte offsets
constexpr size_t OFF_CTRL = 0;                                     // cnt u32 @0, nsel f32 @4
constexpr size_t OFF_WIH  = 256;                                   // bf16 [3072*256]
constexpr size_t OFF_WHH  = OFF_WIH  + (size_t)kN1 * kF * 2;       // bf16 [3072*1024]
constexpr size_t OFF_WOUT = OFF_WHH  + (size_t)kN1 * kH * 2;       // bf16 [256*1024]
constexpr size_t OFF_WSL  = OFF_WOUT + (size_t)kOUT * kH * 2;      // bf16 [256*1024]
constexpr size_t OFF_WX   = OFF_WSL  + (size_t)kF * kH * 2;        // bf16 [64*256]
constexpr size_t OFF_HBF  = OFF_WX   + (size_t)kB * kF * 2;        // bf16 [64*1024]
constexpr size_t OFF_HF   = OFF_HBF  + (size_t)kB * kH * 2;        // f32  [64*1024]
constexpr size_t OFF_GI   = OFF_HF   + (size_t)kB * kH * 4;        // f32  [64*3072]
constexpr size_t OFF_GH   = OFF_GI   + (size_t)kB * kN1 * 4;       // f32  [64*3072]
constexpr size_t OFF_LG   = OFF_GH   + (size_t)kB * kN1 * 4;       // f32  [64*256]

__device__ __forceinline__ unsigned short f2bf(float f) {
  unsigned u = __builtin_bit_cast(unsigned, f);
  u += 0x7FFFu + ((u >> 16) & 1u);                // round-to-nearest-even
  return (unsigned short)(u >> 16);
}

// A fragment: 16x32 bf16, row-major source [.. x ld].
// Contiguous-k layout used here: lane<16 -> M=lane, K=0..15; lane>=16 -> M=lane-16, K=16..31
__device__ __forceinline__ v16bf load_a(const unsigned short* __restrict__ p,
                                        int ld, int m0, int k0, int lane) {
  union { v16bf v; unsigned u[8]; } r;
  const int row   = m0 + (lane & 15);
  const int half16 = (lane < 16) ? 0 : 16;
  const unsigned* q = (const unsigned*)(p + (size_t)row * ld + k0 + half16);
#pragma unroll
  for (int i = 0; i < 8; ++i) r.u[i] = q[i];
  return r.v;
}

// B fragment: 32x16 bf16 of W^T where W is row-major [N x ld] (ld = K).
// Contiguous-k layout used here (same as load_a): N = lane%16; lane<16 -> K=0..15, lane>=16 -> K=16..31 (contiguous in W's row)
__device__ __forceinline__ v16bf load_b(const unsigned short* __restrict__ w,
                                        int ld, int n0, int k0, int lane) {
  union { v16bf v; unsigned u[8]; } r;
  const int n      = n0 + (lane & 15);
  const int half16 = (lane < 16) ? 0 : 16;
  const unsigned* q = (const unsigned*)(w + (size_t)n * ld + k0 + half16);
#pragma unroll
  for (int i = 0; i < 8; ++i) r.u[i] = q[i];
  return r.v;
}

__device__ __forceinline__ v8f wmma_bf16(v16bf a, v16bf b, v8f c) {
  return __builtin_amdgcn_wmma_f32_16x16x32_bf16(false, a, false, b, (short)0, c, false, false);
}

// D/C f32 16x16: lane<16 -> M=m0+j, lane>=16 -> M=m0+8+j; N=n0+lane%16
__device__ __forceinline__ void store_d(float* __restrict__ d, int ld, int m0, int n0,
                                        int lane, v8f acc) {
  const int n  = n0 + (lane & 15);
  const int mb = m0 + ((lane < 16) ? 0 : 8);
#pragma unroll
  for (int j = 0; j < 8; ++j) d[(size_t)(mb + j) * ld + n] = acc[j];
}

__device__ __forceinline__ void store_d_bias(float* __restrict__ d, int ld, int m0, int n0,
                                             int lane, v8f acc, const float* __restrict__ bias) {
  const int n  = n0 + (lane & 15);
  const int mb = m0 + ((lane < 16) ? 0 : 8);
  const float bv = bias[n];
#pragma unroll
  for (int j = 0; j < 8; ++j) d[(size_t)(mb + j) * ld + n] = acc[j] + bv;
}

__device__ __forceinline__ void grid_sync(unsigned* cnt, unsigned& bars) {
  __syncthreads();
  ++bars;
  if (threadIdx.x == 0) {
    __threadfence();  // release our global writes to device scope
    __hip_atomic_fetch_add(cnt, 1u, __ATOMIC_RELEASE, __HIP_MEMORY_SCOPE_AGENT);
    const unsigned target = bars * (unsigned)NBLK;
    while (__hip_atomic_load(cnt, __ATOMIC_ACQUIRE, __HIP_MEMORY_SCOPE_AGENT) < target) {
      __builtin_amdgcn_s_sleep(8);
    }
  }
  __syncthreads();
  __threadfence();    // acquire: don't read stale lines written by other WGPs
}

__device__ __forceinline__ float sigmoidf_(float x) { return 1.0f / (1.0f + expf(-x)); }

} // namespace

__global__ void convert_weights_kernel(const float* __restrict__ Wih, const float* __restrict__ Whh,
                                       const float* __restrict__ Wout, const float* __restrict__ Wsel,
                                       unsigned short* __restrict__ dWih, unsigned short* __restrict__ dWhh,
                                       unsigned short* __restrict__ dWout, unsigned short* __restrict__ dWsel) {
  const int S0 = kN1 * kF, S1 = kN1 * kH, S2 = kOUT * kH, S3 = kF * kH;
  const int total = S0 + S1 + S2 + S3;
  for (int i = blockIdx.x * blockDim.x + threadIdx.x; i < total; i += gridDim.x * blockDim.x) {
    if (i < S0)                dWih[i] = f2bf(Wih[i]);
    else if (i < S0 + S1)      dWhh[i - S0] = f2bf(Whh[i - S0]);
    else if (i < S0 + S1 + S2) dWout[i - S0 - S1] = f2bf(Wout[i - S0 - S1]);
    else                       dWsel[i - S0 - S1 - S2] = f2bf(Wsel[i - S0 - S1 - S2]);
  }
}

__global__ __launch_bounds__(NTHR, 1) void adaptive_gru_persistent(
    const float* __restrict__ x, const float* __restrict__ noise,
    const float* __restrict__ bih, const float* __restrict__ bhh,
    const float* __restrict__ bout, const float* __restrict__ bsel,
    const unsigned short* __restrict__ Wih_bf, const unsigned short* __restrict__ Whh_bf,
    const unsigned short* __restrict__ Wout_bf, const unsigned short* __restrict__ Wsel_bf,
    unsigned short* __restrict__ wx_bf, unsigned short* __restrict__ h_bf,
    float* __restrict__ h_f32, float* __restrict__ gi, float* __restrict__ gh,
    float* __restrict__ logits, unsigned* cnt, float* nsel, float* __restrict__ out) {
  const int gtid = blockIdx.x * NTHR + threadIdx.x;
  const int wave = gtid >> 5;
  const int lane = threadIdx.x & 31;
  unsigned bars = 0;

  // ---- init: h=0, wx = bf16(x[:,0,:]), weights[0]=1 ----
  for (int i = gtid; i < kB * kH; i += NBLK * NTHR) {
    h_f32[i] = 0.0f;
    h_bf[i]  = 0;
  }
  if (gtid < kB * kF) {
    const int b = gtid >> 8, f = gtid & (kF - 1);
    out[WSEL_OUT_OFF + gtid] = 1.0f;                      // w0 == ones
    wx_bf[gtid] = f2bf(x[(size_t)b * kT * kF + f]);       // x[b,0,f]
  }
  grid_sync(cnt, bars);

  for (int t = 0; t < kT; ++t) {
    // ---- Phase A: gi = wx @ Wih^T  and  gh = h @ Whh^T ----
    {
      const int ct = wave % (kN1 / 16);   // 0..191
      const int rt = wave / (kN1 / 16);   // 0..3
      const int m0 = rt * 16, n0 = ct * 16;
      v8f acc = {};
      for (int k0 = 0; k0 < kH; k0 += 32)
        acc = wmma_bf16(load_a(h_bf, kH, m0, k0, lane),
                        load_b(Whh_bf, kH, n0, k0, lane), acc);
      store_d(gh, kN1, m0, n0, lane, acc);
      v8f acc2 = {};
      for (int k0 = 0; k0 < kF; k0 += 32)
        acc2 = wmma_bf16(load_a(wx_bf, kF, m0, k0, lane),
                         load_b(Wih_bf, kF, n0, k0, lane), acc2);
      store_d(gi, kN1, m0, n0, lane, acc2);
    }
    grid_sync(cnt, bars);

    // ---- Phase B: gate nonlinearities, h update ----
    for (int i = gtid; i < kB * kH; i += NBLK * NTHR) {
      const int b = i >> 10, j = i & (kH - 1);
      const size_t base = (size_t)b * kN1;
      const float i_r = gi[base + j]          + bih[j];
      const float i_z = gi[base + kH + j]     + bih[kH + j];
      const float i_n = gi[base + 2 * kH + j] + bih[2 * kH + j];
      const float h_r = gh[base + j]          + bhh[j];
      const float h_z = gh[base + kH + j]     + bhh[kH + j];
      const float h_n = gh[base + 2 * kH + j] + bhh[2 * kH + j];
      const float r = sigmoidf_(i_r + h_r);
      const float z = sigmoidf_(i_z + h_z);
      const float n = tanhf(i_n + r * h_n);
      const float hn = (1.0f - z) * n + z * h_f32[i];
      h_f32[i] = hn;
      h_bf[i]  = f2bf(hn);
    }
    grid_sync(cnt, bars);

    // ---- Phase C: out = h @ Wout^T + bout ; logits = h @ Wsel^T + bsel ----
    if (wave < 128) {
      const int ct = wave & 31;           // 0..31 (first 16 -> Wout, rest -> Wsel)
      const int rt = wave >> 5;           // 0..3
      const int m0 = rt * 16;
      const int n0 = (ct & 15) * 16;
      const unsigned short* Wm = (ct < 16) ? Wout_bf : Wsel_bf;
      v8f acc = {};
      for (int k0 = 0; k0 < kH; k0 += 32)
        acc = wmma_bf16(load_a(h_bf, kH, m0, k0, lane),
                        load_b(Wm, kH, n0, k0, lane), acc);
      if (ct < 16) store_d_bias(out + (size_t)t * kB * kOUT, kOUT, m0, n0, lane, acc, bout);
      else         store_d_bias(logits, kF, m0, n0, lane, acc, bsel);
    }
    grid_sync(cnt, bars);

    // ---- Phase D: next gumbel-sigmoid gate, wx, nsel ----
    if (t < kT - 1) {
      float wsum = 0.0f;
      if (gtid < kB * kF) {
        const int b = gtid >> 8, f = gtid & (kF - 1);
        const float nt = noise[(size_t)t * kB * kF + gtid];
        const float y  = logits[gtid] + logf(nt + kEPS) - logf(1.0f - nt + kEPS);
        const float w  = sigmoidf_(y / kTEMP);
        wsum = w;
        out[WSEL_OUT_OFF + (size_t)(t + 1) * kB * kF + gtid] = w;
        const float xv = x[(size_t)b * kT * kF + (size_t)(t + 1) * kF + f];
        wx_bf[gtid] = f2bf(w * xv);
      }
#pragma unroll
      for (int o = 16; o > 0; o >>= 1) wsum += __shfl_xor(wsum, o, 32);
      if (lane == 0) atomicAdd(nsel, wsum);   // global_atomic_add_f32
      grid_sync(cnt, bars);
    }
  }

  grid_sync(cnt, bars);
  if (blockIdx.x == 0 && threadIdx.x == 0)
    out[NSEL_OFF] = __hip_atomic_load(nsel, __ATOMIC_ACQUIRE, __HIP_MEMORY_SCOPE_AGENT);
}

extern "C" void kernel_launch(void* const* d_in, const int* in_sizes, int n_in,
                              void* d_out, int out_size, void* d_ws, size_t ws_size,
                              hipStream_t stream) {
  (void)in_sizes; (void)n_in; (void)out_size; (void)ws_size;
  const float* x     = (const float*)d_in[0];
  const float* noise = (const float*)d_in[1];
  const float* Wih   = (const float*)d_in[2];
  const float* Whh   = (const float*)d_in[3];
  const float* bih   = (const float*)d_in[4];
  const float* bhh   = (const float*)d_in[5];
  const float* Wout  = (const float*)d_in[6];
  const float* bout  = (const float*)d_in[7];
  const float* Wsel  = (const float*)d_in[8];
  const float* bsel  = (const float*)d_in[9];

  char* ws = (char*)d_ws;
  unsigned*       cnt     = (unsigned*)(ws + OFF_CTRL);
  float*          nsel    = (float*)(ws + OFF_CTRL + 4);
  unsigned short* Wih_bf  = (unsigned short*)(ws + OFF_WIH);
  unsigned short* Whh_bf  = (unsigned short*)(ws + OFF_WHH);
  unsigned short* Wout_bf = (unsigned short*)(ws + OFF_WOUT);
  unsigned short* Wsel_bf = (unsigned short*)(ws + OFF_WSL);
  unsigned short* wx_bf   = (unsigned short*)(ws + OFF_WX);
  unsigned short* h_bf    = (unsigned short*)(ws + OFF_HBF);
  float*          h_f32   = (float*)(ws + OFF_HF);
  float*          gi      = (float*)(ws + OFF_GI);
  float*          gh      = (float*)(ws + OFF_GH);
  float*          logits  = (float*)(ws + OFF_LG);

  hipMemsetAsync(d_ws, 0, 256, stream);   // reset barrier counter + nsel each call

  convert_weights_kernel<<<512, 256, 0, stream>>>(Wih, Whh, Wout, Wsel,
                                                  Wih_bf, Whh_bf, Wout_bf, Wsel_bf);

  adaptive_gru_persistent<<<NBLK, NTHR, 0, stream>>>(
      x, noise, bih, bhh, bout, bsel,
      Wih_bf, Whh_bf, Wout_bf, Wsel_bf,
      wx_bf, h_bf, h_f32, gi, gh, logits,
      cnt, nsel, (float*)d_out);
}
